// DiffnetPPModel_89249420411229
// MI455X (gfx1250) — matrix operand order repaired and checked by reading, not compiled, not run on hardware
//
#include <hip/hip_runtime.h>
#include <math.h>

// ---------------------------------------------------------------------------
// DiffNet++ forward for MI455X (gfx1250, wave32).
//  - Dense 64x64 feature transforms: bf16 WMMA (v_wmma_f32_16x16x32_bf16),
//    one wave = 16x64 output stripe = 8 WMMAs, weights staged in LDS.
//    Native __bf16 casts -> hardware cvt; stores use constant-offset base.
//  - GATv2 segment softmax: 3 streaming edge passes with f32 atomics
//    (accumulators fit in the 192MB L2).
//  - Gate MLP collapsed algebraically: (h@W1+b1)@W2+b2 == h.(W1@W2) + const.
// ---------------------------------------------------------------------------

typedef __attribute__((ext_vector_type(16))) __bf16 v16bf;
typedef __attribute__((ext_vector_type(8)))  float  v8f;

__device__ __forceinline__ void atomicMaxF(float* addr, float val) {
  if (val >= 0.0f) atomicMax((int*)addr, __float_as_int(val));
  else             atomicMin((unsigned int*)addr, __float_as_uint(val));
}

// ---------------------------------------------------------------------------
// Y[M,64] = X[M,64] @ W[64,64] + b   (bf16 WMMA, f32 accumulate)
// ---------------------------------------------------------------------------
__global__ __launch_bounds__(256)
void gemm64_wmma(const float* __restrict__ X, const float* __restrict__ W,
                 const float* __restrict__ Bv, float* __restrict__ Y, int M) {
  __shared__ float sW[64 * 64];
  __shared__ float sB[64];
  for (int i = threadIdx.x; i < 64 * 64; i += 256) sW[i] = W[i];
  if (threadIdx.x < 64) sB[threadIdx.x] = Bv[threadIdx.x];
  __syncthreads();

  const int lane   = threadIdx.x & 31;
  const int wave   = threadIdx.x >> 5;      // 8 waves / block
  const int grp    = lane >> 4;             // lane group 0 / 1
  const int lane16 = lane & 15;

  // B fragments (32x16 bf16): lane = column N (mod 16); element j -> K = k0 + grp*16 + j
  v16bf bfrag[4][2];
#pragma unroll
  for (int t = 0; t < 4; ++t) {
#pragma unroll
    for (int s = 0; s < 2; ++s) {
      v16bf tmp;
      const int n  = t * 16 + lane16;
      const int kb = s * 32 + grp * 16;
#pragma unroll
      for (int j = 0; j < 16; ++j) tmp[j] = (__bf16)sW[(kb + j) * 64 + n];
      bfrag[t][s] = tmp;
    }
  }

  const int ntiles = (M + 15) >> 4;
  for (int tile = blockIdx.x * 8 + wave; tile < ntiles; tile += gridDim.x * 8) {
    const int m0  = tile << 4;
    const int row = m0 + lane16;
    const float* Xr = X + (size_t)row * 64;
    // A fragments (16x32 bf16): row = lane16; element j -> K = kb + j (kb = s*32 + grp*16)
    v16bf afrag[2];
#pragma unroll
    for (int s = 0; s < 2; ++s) {
      const int kb = s * 32 + grp * 16;
      const float4 a0 = *(const float4*)(Xr + kb);
      const float4 a1 = *(const float4*)(Xr + kb + 4);
      const float4 b0 = *(const float4*)(Xr + kb + 8);
      const float4 b1 = *(const float4*)(Xr + kb + 12);
      v16bf tmp;
      tmp[0]  = (__bf16)a0.x; tmp[1]  = (__bf16)a0.y;
      tmp[2]  = (__bf16)a0.z; tmp[3]  = (__bf16)a0.w;
      tmp[4]  = (__bf16)a1.x; tmp[5]  = (__bf16)a1.y;
      tmp[6]  = (__bf16)a1.z; tmp[7]  = (__bf16)a1.w;
      tmp[8]  = (__bf16)b0.x; tmp[9]  = (__bf16)b0.y;
      tmp[10] = (__bf16)b0.z; tmp[11] = (__bf16)b0.w;
      tmp[12] = (__bf16)b1.x; tmp[13] = (__bf16)b1.y;
      tmp[14] = (__bf16)b1.z; tmp[15] = (__bf16)b1.w;
      afrag[s] = tmp;
    }
    v8f acc[4] = {};
#pragma unroll
    for (int t = 0; t < 4; ++t) {
      acc[t] = __builtin_amdgcn_wmma_f32_16x16x32_bf16(false, afrag[0], false, bfrag[t][0],
                                                       (short)0, acc[t], false, false);
      acc[t] = __builtin_amdgcn_wmma_f32_16x16x32_bf16(false, afrag[1], false, bfrag[t][1],
                                                       (short)0, acc[t], false, false);
    }
    // D layout: VGPR r -> row m0 + grp*8 + r, col = t*16 + lane16
    float* Ybase = Y + (size_t)(m0 + grp * 8) * 64;
#pragma unroll
    for (int t = 0; t < 4; ++t) {
      const int col  = t * 16 + lane16;
      const float bb = sB[col];
      float* Yr = Ybase + col;
#pragma unroll
      for (int r = 0; r < 8; ++r) {
        Yr[r * 64] = acc[t][r] + bb;      // constant 256B-stride offsets
      }
    }
  }
}

// ---------------------------------------------------------------------------
// misc small kernels
// ---------------------------------------------------------------------------
__global__ void fill_kernel(float* p, float v, int n) {
  int i = blockIdx.x * blockDim.x + threadIdx.x;
  if (i < n) p[i] = v;
}

// out[i] = bias[i%64] (+ res[i])  -- pre-seed GAT accumulator with bias/residual
__global__ void gat_init_out(float* __restrict__ out, const float* __restrict__ bias,
                             const float* __restrict__ res, int n64) {
  int i = blockIdx.x * blockDim.x + threadIdx.x;
  if (i >= n64) return;
  float v = bias[i & 63];
  if (res) v += res[i];
  out[i] = v;
}

// pass 1: e = attn . leaky_relu(fs[src]+fd[dst], 0.2); atomic max into m[dst]
__global__ void edge_logits(const float* __restrict__ fs, const float* __restrict__ fd,
                            const int* __restrict__ src, const int* __restrict__ dst,
                            const float* __restrict__ attn, float* __restrict__ ebuf,
                            float* __restrict__ m, int nE) {
  __shared__ float sa[64];
  if (threadIdx.x < 64) sa[threadIdx.x] = attn[threadIdx.x];
  __syncthreads();
  int e = blockIdx.x * blockDim.x + threadIdx.x;
  if (e >= nE) return;
  const float* a = fs + (size_t)src[e] * 64;
  const float* b = fd + (size_t)dst[e] * 64;
  float acc = 0.f;
#pragma unroll 4
  for (int k = 0; k < 64; k += 4) {
    float4 x = *(const float4*)(a + k);
    float4 y = *(const float4*)(b + k);
    float s0 = x.x + y.x; s0 = s0 > 0.f ? s0 : 0.2f * s0;
    float s1 = x.y + y.y; s1 = s1 > 0.f ? s1 : 0.2f * s1;
    float s2 = x.z + y.z; s2 = s2 > 0.f ? s2 : 0.2f * s2;
    float s3 = x.w + y.w; s3 = s3 > 0.f ? s3 : 0.2f * s3;
    acc += s0 * sa[k] + s1 * sa[k + 1] + s2 * sa[k + 2] + s3 * sa[k + 3];
  }
  ebuf[e] = acc;
  atomicMaxF(m + dst[e], acc);
}

// pass 2: ex = exp(e - m[dst]); atomic add denom[dst]
__global__ void edge_exp(const int* __restrict__ dst, float* __restrict__ ebuf,
                         const float* __restrict__ m, float* __restrict__ denom, int nE) {
  int e = blockIdx.x * blockDim.x + threadIdx.x;
  if (e >= nE) return;
  int d = dst[e];
  float ex = __expf(ebuf[e] - m[d]);
  ebuf[e] = ex;
  atomicAdd(denom + d, ex);
}

// pass 3 (wave per edge): out[dst] += (ex/denom[dst]) * fs[src]
__global__ void edge_scatter(const float* __restrict__ fs, const float* __restrict__ ebuf,
                             const float* __restrict__ denom, const int* __restrict__ src,
                             const int* __restrict__ dst, float* __restrict__ out, int nE) {
  int wid  = (blockIdx.x * blockDim.x + threadIdx.x) >> 5;
  int lane = threadIdx.x & 31;
  int nw   = (gridDim.x * blockDim.x) >> 5;
  for (int e = wid; e < nE; e += nw) {
    int s = src[e], d = dst[e];
    float alpha = ebuf[e] / denom[d];
    const float* fr = fs + (size_t)s * 64;
    float* o = out + (size_t)d * 64;
    atomicAdd(o + lane,       alpha * fr[lane]);
    atomicAdd(o + lane + 32,  alpha * fr[lane + 32]);
  }
}

// collapse gate MLP: v[r] = sum_j W1[r][j]*W2[j]; c = b1.W2 + b2
__global__ void att_precompute(const float* __restrict__ W1, const float* __restrict__ b1,
                               const float* __restrict__ W2, const float* __restrict__ b2,
                               float* __restrict__ v, float* __restrict__ c) {
  int r = threadIdx.x;
  if (r < 128) {
    float acc = 0.f;
    for (int j = 0; j < 128; ++j) acc += W1[r * 128 + j] * W2[j];
    v[r] = acc;
  }
  if (r == 0) {
    float acc = b2[0];
    for (int j = 0; j < 128; ++j) acc += b1[j] * W2[j];
    c[0] = acc;
  }
}

// z_inf/z_int per user + block-reduced batch stats for BN
__global__ __launch_bounds__(256)
void att_z_kernel(const float* __restrict__ hu, const float* __restrict__ p,
                  const float* __restrict__ q, const float* __restrict__ vattn,
                  const float* __restrict__ cattn, float* __restrict__ zinf,
                  float* __restrict__ zint, float* __restrict__ stats, int N) {
  int u = blockIdx.x * blockDim.x + threadIdx.x;
  float zi = 0.f, zt = 0.f;
  if (u < N) {
    const float* hr = hu + (size_t)u * 64;
    const float* pr = p  + (size_t)u * 64;
    const float* qr = q  + (size_t)u * 64;
    const float* vi = vattn;
    const float* vt = vattn + 128;
    float a = cattn[0], b = cattn[1];
    for (int k = 0; k < 64; ++k) { float h = hr[k]; a += h * vi[k]; b += h * vt[k]; }
    for (int k = 0; k < 64; ++k) { a += pr[k] * vi[64 + k]; b += qr[k] * vt[64 + k]; }
    zinf[u] = a; zint[u] = b; zi = a; zt = b;
  }
  __shared__ float s0[256], s1[256], s2[256], s3[256];
  int t = threadIdx.x;
  s0[t] = zi; s1[t] = zi * zi; s2[t] = zt; s3[t] = zt * zt;
  __syncthreads();
  for (int sft = 128; sft > 0; sft >>= 1) {
    if (t < sft) { s0[t] += s0[t + sft]; s1[t] += s1[t + sft];
                   s2[t] += s2[t + sft]; s3[t] += s3[t + sft]; }
    __syncthreads();
  }
  if (t == 0) {
    atomicAdd(stats + 0, s0[0]); atomicAdd(stats + 1, s1[0]);
    atomicAdd(stats + 2, s2[0]); atomicAdd(stats + 3, s3[0]);
  }
}

// BN + leaky_relu(0.01) + 2-way softmax -> gammas written back into z buffers
__global__ void att_gamma_kernel(float* __restrict__ zinf, float* __restrict__ zint,
                                 const float* __restrict__ stats, int N) {
  int u = blockIdx.x * blockDim.x + threadIdx.x;
  if (u >= N) return;
  float inv = 1.f / (float)N;
  float mu0 = stats[0] * inv, v0 = stats[1] * inv - mu0 * mu0;
  float mu1 = stats[2] * inv, v1 = stats[3] * inv - mu1 * mu1;
  float a0 = (zinf[u] - mu0) * rsqrtf(v0 + 1e-5f); a0 = a0 > 0.f ? a0 : 0.01f * a0;
  float a1 = (zint[u] - mu1) * rsqrtf(v1 + 1e-5f); a1 = a1 > 0.f ? a1 : 0.01f * a1;
  float mx = fmaxf(a0, a1);
  float e0 = __expf(a0 - mx), e1 = __expf(a1 - mx);
  float s = e0 + e1;
  zinf[u] = e0 / s; zint[u] = e1 / s;
}

// hu_next = g0*p + g1*q + hu
__global__ void combine_kernel(const float* __restrict__ hu, const float* __restrict__ p,
                               const float* __restrict__ q, const float* __restrict__ g0,
                               const float* __restrict__ g1, float* __restrict__ out, int n64) {
  int i = blockIdx.x * blockDim.x + threadIdx.x;
  if (i >= n64) return;
  int u = i >> 6;
  out[i] = g0[u] * p[i] + g1[u] * q[i] + hu[i];
}

// wave-per-pair 192-dim dot over the 3 concat levels; j<EP -> pos, else neg
__global__ void score_kernel(const float* __restrict__ eu, const float* __restrict__ hu1,
                             const float* __restrict__ hu2, const float* __restrict__ ei,
                             const float* __restrict__ hi1, const float* __restrict__ hi2,
                             const int* __restrict__ pu, const int* __restrict__ pi,
                             const int* __restrict__ nu, const int* __restrict__ ni,
                             float* __restrict__ out, int EP) {
  int wid  = (blockIdx.x * blockDim.x + threadIdx.x) >> 5;
  int lane = threadIdx.x & 31;
  int nw   = (gridDim.x * blockDim.x) >> 5;
  for (int j = wid; j < 2 * EP; j += nw) {
    int u  = (j < EP) ? pu[j] : nu[j - EP];
    int it = (j < EP) ? pi[j] : ni[j - EP];
    const float* U[3] = { eu, hu1, hu2 };
    const float* I[3] = { ei, hi1, hi2 };
    float acc = 0.f;
#pragma unroll
    for (int lv = 0; lv < 3; ++lv) {
      const float* ur = U[lv] + (size_t)u  * 64;
      const float* ir = I[lv] + (size_t)it * 64;
      acc += ur[lane] * ir[lane] + ur[lane + 32] * ir[lane + 32];
    }
#pragma unroll
    for (int off = 16; off > 0; off >>= 1) acc += __shfl_down(acc, off, 32);
    if (lane == 0) out[j] = acc;
  }
}

// ---------------------------------------------------------------------------
extern "C" void kernel_launch(void* const* d_in, const int* in_sizes, int n_in,
                              void* d_out, int out_size, void* d_ws, size_t ws_size,
                              hipStream_t stream) {
  const int*   rate_src  = (const int*)d_in[0];
  const int*   rate_dst  = (const int*)d_in[1];
  const int*   trust_src = (const int*)d_in[2];
  const int*   trust_dst = (const int*)d_in[3];
  const int*   pos_u     = (const int*)d_in[4];
  const int*   pos_i     = (const int*)d_in[5];
  const int*   neg_u     = (const int*)d_in[6];
  const int*   neg_i     = (const int*)d_in[7];
  const float* eu        = (const float*)d_in[8];
  const float* ei        = (const float*)d_in[9];
  const float* rate_W    = (const float*)d_in[10];
  const float* rate_b    = (const float*)d_in[11];
  const float* rate_attn = (const float*)d_in[12];
  const float* rate_bias = (const float*)d_in[13];
  const float* rb_W      = (const float*)d_in[14];
  const float* rb_b      = (const float*)d_in[15];
  const float* rb_attn   = (const float*)d_in[16];
  const float* rb_bias   = (const float*)d_in[17];
  const float* tr_W      = (const float*)d_in[18];
  const float* tr_b      = (const float*)d_in[19];
  const float* tr_attn   = (const float*)d_in[20];
  const float* tr_bias   = (const float*)d_in[21];
  const float* attW1     = (const float*)d_in[22];
  const float* attb1     = (const float*)d_in[23];
  const float* attW2     = (const float*)d_in[24];
  const float* attb2     = (const float*)d_in[25];

  const int nE_rate = in_sizes[0];
  const int nE_tr   = in_sizes[2];
  const int nPairs  = in_sizes[4];
  const int NUn     = in_sizes[8] / 64;
  const int NIn     = in_sizes[9] / 64;
  const int L       = 2;

  const size_t NU64 = (size_t)NUn * 64;
  const size_t NI64 = (size_t)NIn * 64;

  float* ws = (float*)d_ws;
  size_t off = 0;
  auto take = [&](size_t n) { float* r = ws + off; off += n; return r; };
  float* hu1   = take(NU64);
  float* hu2   = take(NU64);
  float* hi1   = take(NI64);
  float* hi2   = take(NI64);
  float* fsb   = take(NU64);
  float* fdb   = take(NU64);
  float* pb    = take(NU64);
  float* qb    = take(NU64);
  float* ebuf  = take((size_t)(nE_rate > nE_tr ? nE_rate : nE_tr));
  float* mbuf  = take((size_t)NUn);
  float* dbuf  = take((size_t)NUn);
  float* zinf  = take((size_t)NUn);
  float* zint  = take((size_t)NUn);
  float* stats = take(4);
  float* attv  = take(256);
  float* attc  = take(2);
  (void)ws_size; (void)n_in; (void)out_size;

  auto gemm = [&](const float* X, const float* W, const float* B, float* Y, int M) {
    int blocks = ((M + 15) / 16 + 7) / 8;
    gemm64_wmma<<<blocks, 256, 0, stream>>>(X, W, B, Y, M);
  };
  auto gat = [&](const float* fs, const float* fd, const int* src, const int* dst,
                 const float* attn, const float* bias, const float* res,
                 float* outp, int nE, int Nd) {
    int nd64 = Nd * 64;
    gat_init_out<<<(nd64 + 255) / 256, 256, 0, stream>>>(outp, bias, res, nd64);
    fill_kernel<<<(Nd + 255) / 256, 256, 0, stream>>>(mbuf, -INFINITY, Nd);
    hipMemsetAsync(dbuf, 0, (size_t)Nd * sizeof(float), stream);
    edge_logits<<<(nE + 255) / 256, 256, 0, stream>>>(fs, fd, src, dst, attn, ebuf, mbuf, nE);
    edge_exp<<<(nE + 255) / 256, 256, 0, stream>>>(dst, ebuf, mbuf, dbuf, nE);
    edge_scatter<<<(nE + 7) / 8, 256, 0, stream>>>(fs, ebuf, dbuf, src, dst, outp, nE);
  };

  const float* hu_cur = eu;
  const float* hi_cur = ei;
  for (int l = 0; l < L; ++l) {
    float* hu_next = (l == 0) ? hu1 : hu2;
    float* hi_next = (l == 0) ? hi1 : hi2;

    for (int i = 0; i < 2; ++i)
      att_precompute<<<1, 128, 0, stream>>>(attW1 + (size_t)(l * 2 + i) * 128 * 128,
                                            attb1 + (l * 2 + i) * 128,
                                            attW2 + (l * 2 + i) * 128,
                                            attb2 + (l * 2 + i),
                                            attv + i * 128, attc + i);

    // rate GAT: user -> item, residual +hi, out = hi_next
    gemm(hu_cur, rate_W + (size_t)(l * 2 + 0) * 4096, rate_b + (l * 2 + 0) * 64, fsb, NUn);
    gemm(hi_cur, rate_W + (size_t)(l * 2 + 1) * 4096, rate_b + (l * 2 + 1) * 64, fdb, NIn);
    gat(fsb, fdb, rate_src, rate_dst, rate_attn + l * 64, rate_bias + l * 64,
        hi_cur, hi_next, nE_rate, NIn);

    // rated-by GAT: item -> user (reversed edges), out = q
    gemm(hi_cur, rb_W + (size_t)(l * 2 + 0) * 4096, rb_b + (l * 2 + 0) * 64, fsb, NIn);
    gemm(hu_cur, rb_W + (size_t)(l * 2 + 1) * 4096, rb_b + (l * 2 + 1) * 64, fdb, NUn);
    gat(fsb, fdb, rate_dst, rate_src, rb_attn + l * 64, rb_bias + l * 64,
        nullptr, qb, nE_rate, NUn);

    // trust GAT: user -> user, out = p
    gemm(hu_cur, tr_W + (size_t)(l * 2 + 0) * 4096, tr_b + (l * 2 + 0) * 64, fsb, NUn);
    gemm(hu_cur, tr_W + (size_t)(l * 2 + 1) * 4096, tr_b + (l * 2 + 1) * 64, fdb, NUn);
    gat(fsb, fdb, trust_src, trust_dst, tr_attn + l * 64, tr_bias + l * 64,
        nullptr, pb, nE_tr, NUn);

    // gates: z -> BN -> leaky -> softmax -> combine
    hipMemsetAsync(stats, 0, 4 * sizeof(float), stream);
    att_z_kernel<<<(NUn + 255) / 256, 256, 0, stream>>>(hu_cur, pb, qb, attv, attc,
                                                        zinf, zint, stats, NUn);
    att_gamma_kernel<<<(NUn + 255) / 256, 256, 0, stream>>>(zinf, zint, stats, NUn);
    combine_kernel<<<((int)NU64 + 255) / 256, 256, 0, stream>>>(hu_cur, pb, qb, zinf, zint,
                                                                hu_next, (int)NU64);
    hu_cur = hu_next;
    hi_cur = hi_next;
  }

  score_kernel<<<(2 * nPairs + 7) / 8, 256, 0, stream>>>(eu, hu1, hu2, ei, hi1, hi2,
                                                         pos_u, pos_i, neg_u, neg_i,
                                                         (float*)d_out, nPairs);
}
